// TransducerPredictor_76742475645266
// MI455X (gfx1250) — matrix operand order repaired and checked by reading, not compiled, not run on hardware
//
#include <hip/hip_runtime.h>

// ---------------------------------------------------------------------------
// TransducerPredictor: 2-layer LayerNorm-LSTM on gfx1250 (MI455X).
//   N=2 layers, B=64, U=1024 (T=1025 with SOS), H=640, 4H=2560.
// Strategy:
//   - time-parallel input GEMM zx = LN(x@W)+bias via v_wmma_f32_16x16x32_bf16
//   - persistent recurrence kernel: 4 WGs x 16 batch rows, gates + cell state
//     kept in registers (gate columns co-located per lane), h in LDS (bf16).
// ---------------------------------------------------------------------------

#define NLAYERS 2
#define BATCH   64
#define SEQU    1024
#define TT      1025          // U + 1 (SOS prepended)
#define HID     640
#define H4      2560
#define LN_EPS  1e-3f

typedef __bf16 bf16;
typedef __attribute__((ext_vector_type(16))) __bf16 v16bf;
typedef __attribute__((ext_vector_type(8)))  __bf16 v8bf;
typedef __attribute__((ext_vector_type(8)))  float  v8f;

__device__ __forceinline__ bf16 f2bf(float f) {
  union { float f; unsigned u; } in; in.f = f;
  unsigned r = in.u + 0x7FFFu + ((in.u >> 16) & 1u);   // round-to-nearest-even
  union { unsigned short s; bf16 b; } out; out.s = (unsigned short)(r >> 16);
  return out.b;
}
__device__ __forceinline__ float sigm(float x) { return 1.0f / (1.0f + __expf(-x)); }

// ---------------------------------------------------------------------------
// Transpose f32 weights [l][k][n] -> bf16 col-major [l][n][k] so WMMA B
// fragments are contiguous along K (two b128 loads per fragment).
// ---------------------------------------------------------------------------
__global__ void prep_weights_kernel(const float* __restrict__ W,
                                    const float* __restrict__ R,
                                    bf16* __restrict__ Wt,
                                    bf16* __restrict__ Rt) {
  size_t i = (size_t)blockIdx.x * blockDim.x + threadIdx.x;
  if (i >= (size_t)NLAYERS * H4 * HID) return;
  int k = (int)(i % HID);
  int n = (int)((i / HID) % H4);
  int l = (int)(i / ((size_t)HID * H4));
  size_t src = (size_t)l * HID * H4 + (size_t)k * H4 + n;
  Wt[i] = f2bf(W[src]);
  Rt[i] = f2bf(R[src]);
}

// ---------------------------------------------------------------------------
// Embedding gather + SOS row of zeros, output bf16 x[b][t][h].
// ---------------------------------------------------------------------------
__global__ void embed_kernel(const int* __restrict__ targets,
                             const float* __restrict__ table,
                             bf16* __restrict__ x) {
  size_t i = (size_t)blockIdx.x * blockDim.x + threadIdx.x;
  if (i >= (size_t)BATCH * TT * HID) return;
  int h = (int)(i % HID);
  int t = (int)((i / HID) % TT);
  int b = (int)(i / ((size_t)HID * TT));
  float v = 0.0f;
  if (t > 0) {
    int tok = targets[(size_t)b * SEQU + (t - 1)];
    v = table[(size_t)tok * HID + h];
  }
  x[i] = f2bf(v);
}

// ---------------------------------------------------------------------------
// zx_raw = A[M x 640] (bf16 row-major) @ Bt[2560 x 640] (bf16, N-major) -> f32.
// WG = 256 threads (8 waves), tile 64(M) x 128(N), K-step 32, LDS staged.
// ---------------------------------------------------------------------------
__global__ __launch_bounds__(256) void gemm_xw_kernel(const bf16* __restrict__ A,
                                                      const bf16* __restrict__ Bm,
                                                      float* __restrict__ C) {
  __shared__ bf16 As[64][40];    // 64x32 used, +8 pad vs bank conflicts
  __shared__ bf16 Bs[128][40];   // 128x32 used (N-major)

  const int tid  = threadIdx.x;
  const int wid  = tid >> 5, lane = tid & 31;
  const int wm   = wid & 1,  wn   = wid >> 1;
  const int half = lane >> 4, l16 = lane & 15;
  const int m0   = blockIdx.y * 64;
  const int n0   = blockIdx.x * 128;

  v8f acc[2][2] = {};
  for (int k0 = 0; k0 < HID; k0 += 32) {
    {
      int ra = tid >> 2, ca = (tid & 3) * 8;
      *(v8bf*)&As[ra][ca] = *(const v8bf*)&A[(size_t)(m0 + ra) * HID + k0 + ca];
      int rb_ = tid >> 1, cb = (tid & 1) * 16;
      *(v8bf*)&Bs[rb_][cb]     = *(const v8bf*)&Bm[(size_t)(n0 + rb_) * HID + k0 + cb];
      *(v8bf*)&Bs[rb_][cb + 8] = *(const v8bf*)&Bm[(size_t)(n0 + rb_) * HID + k0 + cb + 8];
    }
    __syncthreads();
    const int kbA = half * 8;     // A: elems 0..7 -> K kb..kb+7, 8..15 -> kb+16..
    const int kbB = half * 8;     // B: elems 0..7 -> K kb..kb+7, 8..15 -> kb+16..
    v16bf afrag[2], bfrag[2];
#pragma unroll
    for (int i = 0; i < 2; ++i) {
      int row = wm * 32 + i * 16 + l16;
      ((v8bf*)&afrag[i])[0] = *(const v8bf*)&As[row][kbA];
      ((v8bf*)&afrag[i])[1] = *(const v8bf*)&As[row][kbA + 16];
      int col = wn * 32 + i * 16 + l16;
      ((v8bf*)&bfrag[i])[0] = *(const v8bf*)&Bs[col][kbB];
      ((v8bf*)&bfrag[i])[1] = *(const v8bf*)&Bs[col][kbB + 16];
    }
#pragma unroll
    for (int i = 0; i < 2; ++i)
#pragma unroll
      for (int j = 0; j < 2; ++j)
        acc[i][j] = __builtin_amdgcn_wmma_f32_16x16x32_bf16(
            false, afrag[i], false, bfrag[j], (short)0, acc[i][j], false, false);
    __syncthreads();
  }
#pragma unroll
  for (int i = 0; i < 2; ++i)
#pragma unroll
    for (int j = 0; j < 2; ++j)
#pragma unroll
      for (int e = 0; e < 8; ++e) {
        int row = m0 + wm * 32 + i * 16 + half * 8 + e;
        int col = n0 + wn * 32 + j * 16 + l16;
        C[(size_t)row * H4 + col] = acc[i][j][e];
      }
}

// ---------------------------------------------------------------------------
// In-place row LayerNorm over 2560 + gamma/beta + bias.  One WG per row.
// ---------------------------------------------------------------------------
__global__ __launch_bounds__(256) void ln_rows_kernel(float* __restrict__ Z,
                                                      const float* __restrict__ g,
                                                      const float* __restrict__ be,
                                                      const float* __restrict__ bias) {
  const int tid = threadIdx.x;
  const size_t base = (size_t)blockIdx.x * H4;
  float v[10]; float s = 0.f, q = 0.f;
#pragma unroll
  for (int j = 0; j < 10; ++j) {
    int c = tid + j * 256;
    float x = Z[base + c];
    v[j] = x; s += x; q += x * x;
  }
  __shared__ float ss[256], sq[256];
  ss[tid] = s; sq[tid] = q; __syncthreads();
  for (int off = 128; off > 0; off >>= 1) {
    if (tid < off) { ss[tid] += ss[tid + off]; sq[tid] += sq[tid + off]; }
    __syncthreads();
  }
  float mu   = ss[0] * (1.0f / H4);
  float var  = sq[0] * (1.0f / H4) - mu * mu;
  float rstd = rsqrtf(var + LN_EPS);
#pragma unroll
  for (int j = 0; j < 10; ++j) {
    int c = tid + j * 256;
    Z[base + c] = (v[j] - mu) * rstd * g[c] + be[c] + bias[c];
  }
}

// ---------------------------------------------------------------------------
// Persistent LayerNorm-LSTM recurrence.  Grid = 4 WGs (16 batch rows each),
// 640 threads = 20 waves.  Wave 'wid' owns hidden units [wid*32, wid*32+32)
// across ALL FOUR gates (8 WMMA tiles), so gates + cell state stay in the
// owning lane's registers.  h lives in LDS (bf16) = WMMA A-fragment source.
// ---------------------------------------------------------------------------
__global__ __launch_bounds__(640) void recur_kernel(
    const float* __restrict__ zx,    // [B][T][2560] f32 (LN'ed input preact)
    const bf16*  __restrict__ Rt,    // [2560][640] bf16 (N-major)
    const float* __restrict__ rg, const float* __restrict__ rb,   // [2560]
    const float* __restrict__ sg, const float* __restrict__ sb,   // [640]
    const float* __restrict__ h0, const float* __restrict__ c0,   // [B][640]
    bf16*  __restrict__ hnext,       // [B][T][640] bf16 (next layer in) or null
    float* __restrict__ outp)        // [B][640][T] f32 (final out) or null
{
  __shared__ bf16  hs[16 * HID];                 // 20 KB: current h (bf16)
  __shared__ float wred_s[20 * 16], wred_q[20 * 16];
  __shared__ float cred_s[20 * 16], cred_q[20 * 16];
  __shared__ float stat_mu[16], stat_rs[16];

  const int tid  = threadIdx.x;
  const int wid  = tid >> 5;            // 0..19
  const int lane = tid & 31;
  const int half = lane >> 4, l16 = lane & 15;
  const int b0   = blockIdx.x * 16;

  // tile j: gate g=j>>1, sub=j&1 -> column g*640 + wid*32 + sub*16 + l16
  int   cols[8]; float rgv[8], rbv[8];
#pragma unroll
  for (int j = 0; j < 8; ++j) {
    cols[j] = (j >> 1) * HID + wid * 32 + (j & 1) * 16 + l16;
    rgv[j] = rg[cols[j]]; rbv[j] = rb[cols[j]];
  }
  const int unit0 = wid * 32 + l16;
  float sgv[2] = { sg[unit0], sg[unit0 + 16] };
  float sbv[2] = { sb[unit0], sb[unit0 + 16] };

  // cell state in registers: creg[sub][e] is (row = half*8+e, unit0+sub*16)
  float creg[2][8];
#pragma unroll
  for (int sub = 0; sub < 2; ++sub)
#pragma unroll
    for (int e = 0; e < 8; ++e)
      creg[sub][e] = c0[(size_t)(b0 + half * 8 + e) * HID + unit0 + sub * 16];

  for (int i = tid; i < 16 * HID; i += 640) {
    int r = i / HID, c = i % HID;
    hs[i] = f2bf(h0[(size_t)(b0 + r) * HID + c]);
  }
  __syncthreads();

  for (int t = 0; t < TT; ++t) {
    // -------- r = h @ R : wave computes 16 rows x 128 cols -----------------
    v8f acc[8] = {};
    for (int kc = 0; kc < HID; kc += 32) {
      v16bf a;
      const bf16* hp = &hs[l16 * HID + kc + half * 8];
      ((v8bf*)&a)[0] = *(const v8bf*)hp;
      ((v8bf*)&a)[1] = *(const v8bf*)(hp + 16);
#pragma unroll
      for (int j = 0; j < 8; ++j) {
        v16bf bfr;
        const bf16* bp = &Rt[(size_t)cols[j] * HID + kc + half * 8];
        ((v8bf*)&bfr)[0] = *(const v8bf*)bp;
        ((v8bf*)&bfr)[1] = *(const v8bf*)(bp + 16);
        acc[j] = __builtin_amdgcn_wmma_f32_16x16x32_bf16(
            false, a, false, bfr, (short)0, acc[j], false, false);
      }
    }
    // -------- LayerNorm stats over 2560 per row ---------------------------
    float ps[8], pq[8];
#pragma unroll
    for (int e = 0; e < 8; ++e) { ps[e] = 0.f; pq[e] = 0.f; }
#pragma unroll
    for (int j = 0; j < 8; ++j)
#pragma unroll
      for (int e = 0; e < 8; ++e) { float v = acc[j][e]; ps[e] += v; pq[e] += v * v; }
    for (int m = 1; m < 16; m <<= 1)
#pragma unroll
      for (int e = 0; e < 8; ++e) {
        ps[e] += __shfl_xor(ps[e], m, 32);
        pq[e] += __shfl_xor(pq[e], m, 32);
      }
    if (l16 == 0)
#pragma unroll
      for (int e = 0; e < 8; ++e) {
        wred_s[wid * 16 + half * 8 + e] = ps[e];
        wred_q[wid * 16 + half * 8 + e] = pq[e];
      }
    __syncthreads();
    if (tid < 16) {
      float S = 0.f, Q = 0.f;
      for (int w = 0; w < 20; ++w) { S += wred_s[w * 16 + tid]; Q += wred_q[w * 16 + tid]; }
      float mu  = S * (1.0f / H4);
      float var = Q * (1.0f / H4) - mu * mu;
      stat_mu[tid] = mu; stat_rs[tid] = rsqrtf(var + LN_EPS);
    }
    __syncthreads();
    // -------- z = zx + LN(r)*rg + rb  (in place in acc) -------------------
#pragma unroll
    for (int j = 0; j < 8; ++j)
#pragma unroll
      for (int e = 0; e < 8; ++e) {
        int row = half * 8 + e;
        acc[j][e] = (acc[j][e] - stat_mu[row]) * stat_rs[row] * rgv[j] + rbv[j]
                  + zx[((size_t)(b0 + row) * TT + t) * H4 + cols[j]];
      }
    // -------- gates + cell update (all in-lane) ---------------------------
    float cn[2][8], so[2][8], cps[8], cpq[8];
#pragma unroll
    for (int e = 0; e < 8; ++e) { cps[e] = 0.f; cpq[e] = 0.f; }
#pragma unroll
    for (int sub = 0; sub < 2; ++sub)
#pragma unroll
      for (int e = 0; e < 8; ++e) {
        float zi = acc[0 + sub][e], zf = acc[2 + sub][e];
        float zg = acc[4 + sub][e], zo = acc[6 + sub][e];
        float cc = sigm(zf) * creg[sub][e] + sigm(zi) * tanhf(zg);
        cn[sub][e] = cc; so[sub][e] = sigm(zo);
        cps[e] += cc; cpq[e] += cc * cc;
      }
    for (int m = 1; m < 16; m <<= 1)
#pragma unroll
      for (int e = 0; e < 8; ++e) {
        cps[e] += __shfl_xor(cps[e], m, 32);
        cpq[e] += __shfl_xor(cpq[e], m, 32);
      }
    if (l16 == 0)
#pragma unroll
      for (int e = 0; e < 8; ++e) {
        cred_s[wid * 16 + half * 8 + e] = cps[e];
        cred_q[wid * 16 + half * 8 + e] = cpq[e];
      }
    __syncthreads();
    if (tid < 16) {
      float S = 0.f, Q = 0.f;
      for (int w = 0; w < 20; ++w) { S += cred_s[w * 16 + tid]; Q += cred_q[w * 16 + tid]; }
      float mu  = S * (1.0f / HID);
      float var = Q * (1.0f / HID) - mu * mu;
      stat_mu[tid] = mu; stat_rs[tid] = rsqrtf(var + LN_EPS);
    }
    __syncthreads();
    // -------- c = LN(c); h = sig(zo)*tanh(c); publish h -------------------
#pragma unroll
    for (int sub = 0; sub < 2; ++sub)
#pragma unroll
      for (int e = 0; e < 8; ++e) {
        int row  = half * 8 + e;
        int unit = unit0 + sub * 16;
        float cl = (cn[sub][e] - stat_mu[row]) * stat_rs[row] * sgv[sub] + sbv[sub];
        creg[sub][e] = cl;
        float h = so[sub][e] * tanhf(cl);
        hs[row * HID + unit] = f2bf(h);
        if (hnext) hnext[((size_t)(b0 + row) * TT + t) * HID + unit] = f2bf(h);
        if (outp)  outp[((size_t)(b0 + row) * HID + unit) * TT + t] = h;
      }
    __syncthreads();   // hs stable for next step's WMMA reads
  }
}

// ---------------------------------------------------------------------------
extern "C" void kernel_launch(void* const* d_in, const int* in_sizes, int n_in,
                              void* d_out, int out_size, void* d_ws, size_t ws_size,
                              hipStream_t stream) {
  (void)in_sizes; (void)n_in; (void)out_size; (void)ws_size;
  const int*   targets     = (const int*)d_in[0];
  const float* embed_table = (const float*)d_in[1];
  const float* kernels     = (const float*)d_in[2];
  const float* rec_kernels = (const float*)d_in[3];
  const float* biases      = (const float*)d_in[4];
  const float* kn_gamma    = (const float*)d_in[5];
  const float* kn_beta     = (const float*)d_in[6];
  const float* rn_gamma    = (const float*)d_in[7];
  const float* rn_beta     = (const float*)d_in[8];
  const float* sn_gamma    = (const float*)d_in[9];
  const float* sn_beta     = (const float*)d_in[10];
  const float* h0          = (const float*)d_in[11];
  const float* c0          = (const float*)d_in[12];
  float* out = (float*)d_out;

  char* ws = (char*)d_ws;
  size_t o = 0;
  auto alloc = [&](size_t bytes) -> char* {
    char* p = ws + o;
    o = (o + bytes + 255) & ~(size_t)255;
    return p;
  };
  bf16*  xa = (bf16*)alloc((size_t)BATCH * TT * HID * sizeof(bf16));
  bf16*  xb = (bf16*)alloc((size_t)BATCH * TT * HID * sizeof(bf16));
  float* zx = (float*)alloc((size_t)BATCH * TT * H4 * sizeof(float));
  bf16*  Wt = (bf16*)alloc((size_t)NLAYERS * H4 * HID * sizeof(bf16));
  bf16*  Rt = (bf16*)alloc((size_t)NLAYERS * H4 * HID * sizeof(bf16));

  {
    size_t n = (size_t)NLAYERS * H4 * HID;
    prep_weights_kernel<<<(unsigned)((n + 255) / 256), 256, 0, stream>>>(
        kernels, rec_kernels, Wt, Rt);
  }
  {
    size_t n = (size_t)BATCH * TT * HID;
    embed_kernel<<<(unsigned)((n + 255) / 256), 256, 0, stream>>>(
        targets, embed_table, xa);
  }

  const int M = BATCH * TT;   // 65600
  for (int l = 0; l < NLAYERS; ++l) {
    const bf16* xin = (l == 0) ? xa : xb;
    gemm_xw_kernel<<<dim3(H4 / 128, M / 64), 256, 0, stream>>>(
        xin, Wt + (size_t)l * H4 * HID, zx);
    ln_rows_kernel<<<M, 256, 0, stream>>>(
        zx, kn_gamma + (size_t)l * H4, kn_beta + (size_t)l * H4,
        biases + (size_t)l * H4);
    recur_kernel<<<BATCH / 16, 640, 0, stream>>>(
        zx, Rt + (size_t)l * H4 * HID,
        rn_gamma + (size_t)l * H4, rn_beta + (size_t)l * H4,
        sn_gamma + (size_t)l * HID, sn_beta + (size_t)l * HID,
        h0 + (size_t)l * BATCH * HID, c0 + (size_t)l * BATCH * HID,
        (l == 0) ? xb : (bf16*)nullptr,
        (l == NLAYERS - 1) ? out : (float*)nullptr);
  }
}
